// NeighborhoodAttention2D_17351667875883
// MI455X (gfx1250) — matrix operand order repaired and checked by reading, not compiled, not run on hardware
//
#include <hip/hip_runtime.h>

typedef __attribute__((ext_vector_type(16))) __bf16 v16bf;
typedef __attribute__((ext_vector_type(8)))  __bf16 v8bf;
typedef __attribute__((ext_vector_type(8)))  float  v8f;
typedef __attribute__((ext_vector_type(4)))  unsigned int v4u;
typedef __attribute__((ext_vector_type(8)))  int v8i_;
typedef __attribute__((ext_vector_type(4)))  int v4i_;

#define BB   8
#define CC   128
#define HHT  112
#define WID  112
#define PIX  (HHT*WID)        // 12544
#define NHD  4
#define HD   32
#define KS   7
#define MROWS (BB*PIX)        // 100352

// ---------------------------------------------------------------------------
// TDM helper: 2D tile load Global -> LDS (Tensor Data Mover, gfx1250).
// Descriptor bit layout per CDNA5 ISA 8.3/8.4:
//   g0: [1:0]=count=1, [63:32]=lds_addr, [120:64]=global_addr, [127:126]=type=2
//   g1: data_size(2B)=1 @ [17:16]; tensor_dim0 @ [79:48]; tensor_dim1 @ [111:80];
//       tile_dim0 @ [127:112]; tile_dim1 @ [143:128]; tensor_dim0_stride @ [207:160]
// Dims/strides in data_size units. LDS packing is contiguous (pad disabled).
// ---------------------------------------------------------------------------
#if defined(__has_builtin)
#if __has_builtin(__builtin_amdgcn_tensor_load_to_lds)
#define HAVE_TDM 1
#endif
#endif

__device__ __forceinline__ void tdm_load_2d(unsigned lds_off, const void* gptr,
                                            unsigned tile_d0, unsigned tile_d1,
                                            unsigned tensor_d0, unsigned tensor_d1,
                                            unsigned long long stride0) {
#ifdef HAVE_TDM
    unsigned long long ga = (unsigned long long)gptr;
    v4u g0;
    g0.x = 1u;                                        // count=1 (valid), user mode
    g0.y = lds_off;                                   // lds_addr (bytes)
    g0.z = (unsigned)(ga & 0xFFFFFFFFu);              // global_addr[31:0]
    g0.w = (unsigned)((ga >> 32) & 0x1FFFFFFu) | (2u << 30);   // addr[56:32], type=2
    v8i_ g1;
    g1[0] = (int)(1u << 16);                          // workgroup_mask=0, data_size=2B
    g1[1] = (int)((tensor_d0 & 0xFFFFu) << 16);       // atomic_barrier=0, tdim0 lo
    g1[2] = (int)((tensor_d0 >> 16) | ((tensor_d1 & 0xFFFFu) << 16));
    g1[3] = (int)((tensor_d1 >> 16) | (tile_d0 << 16));
    g1[4] = (int)(tile_d1 & 0xFFFFu);                 // tile_dim1, tile_dim2=0
    g1[5] = (int)(stride0 & 0xFFFFFFFFull);           // tensor_dim0_stride lo
    g1[6] = (int)((stride0 >> 32) & 0xFFFFull);       // stride hi, dim1_stride=0
    g1[7] = 0;
    v4i_ z4 = {0, 0, 0, 0};
#if __clang_major__ >= 23
    v8i_ z8 = {0, 0, 0, 0, 0, 0, 0, 0};
    __builtin_amdgcn_tensor_load_to_lds(g0, g1, z4, z4, z8, 0);
#else
    __builtin_amdgcn_tensor_load_to_lds(g0, g1, z4, z4, 0);
#endif
#else
    (void)lds_off; (void)gptr; (void)tile_d0; (void)tile_d1;
    (void)tensor_d0; (void)tensor_d1; (void)stride0;
#endif
}

__device__ __forceinline__ void tdm_wait() {
#ifdef HAVE_TDM
    __builtin_amdgcn_s_wait_tensorcnt(0);
#endif
}

// ---------------------------------------------------------------------------
// 1) x (B,C,H,W) f32 -> xp (B*H*W, C) bf16, 32x32 LDS tile transpose
// ---------------------------------------------------------------------------
__global__ void transpose_x_kernel(const float* __restrict__ x,
                                   __bf16* __restrict__ xp) {
    __shared__ float tile[32][33];
    const int tid = threadIdx.x;
    int blk = blockIdx.x;
    const int ptiles = PIX / 32;            // 392
    const int ctiles = CC / 32;             // 4
    const int pt = blk % ptiles; blk /= ptiles;
    const int ct = blk % ctiles;
    const int b  = blk / ctiles;
    const int l = tid & 31;
    const int r = tid >> 5;
#pragma unroll
    for (int i = 0; i < 4; ++i) {
        int c = ct * 32 + r + i * 8;
        tile[r + i * 8][l] = x[((size_t)(b * CC + c)) * PIX + pt * 32 + l];
    }
    __syncthreads();
#pragma unroll
    for (int i = 0; i < 4; ++i) {
        int p = pt * 32 + r + i * 8;
        xp[((size_t)b * PIX + p) * CC + ct * 32 + l] = (__bf16)tile[l][r + i * 8];
    }
}

// ---------------------------------------------------------------------------
// 2) weights f32 -> bf16
// ---------------------------------------------------------------------------
__global__ void convert_w_kernel(const float* __restrict__ qkv_w,
                                 const float* __restrict__ proj_w,
                                 __bf16* __restrict__ wqkv,
                                 __bf16* __restrict__ wproj) {
    int i = blockIdx.x * blockDim.x + threadIdx.x;
    if (i < 3 * CC * CC) {
        wqkv[i] = (__bf16)qkv_w[i];
    } else {
        int j = i - 3 * CC * CC;
        if (j < CC * CC) wproj[j] = (__bf16)proj_w[j];
    }
}

// ---------------------------------------------------------------------------
// 3) Fused QKV GEMM + q.k dot + v store.
//    Block = 1 head x 8 waves; wave = 32 pixels. Per-head q/k/v weight slab
//    (3 x 32 x 128 bf16 = 24KB) staged in LDS by one TDM op. B-tiles are
//    consumed per-segment (q, then k, then v) to keep <=2 B-tiles live:
//    peak regs ~ 96 acc + 16 A + 16 B, no spills under launch_bounds(256,1).
// ---------------------------------------------------------------------------
__global__ void __launch_bounds__(256, 1)
qkv_dot_v_kernel(const __bf16* __restrict__ xp,
                 const __bf16* __restrict__ wqkv,
                 const float* __restrict__ qkv_b,
                 __bf16* __restrict__ v_ws,
                 float* __restrict__ dot_ws) {
    __shared__ __align__(64) __bf16 ldsw[3 * HD * CC];   // [s][row-in-head][k]

    const int lane = threadIdx.x & 31;
    const int wv   = threadIdx.x >> 5;      // wave in block
    const int h    = blockIdx.x & (NHD - 1);
    const int mt   = (blockIdx.x >> 2) * 8 + wv;
    const int m0   = mt * 32;
    const int col  = lane & 15;
    const int hseg = lane >> 4;

    // stage per-head weights: 3 segments of 32x128 at stride 128x128 elems
    if (wv == 0) {
#ifdef HAVE_TDM
        tdm_load_2d((unsigned)(uintptr_t)ldsw,
                    wqkv + (size_t)h * HD * CC,
                    /*tile_d0=*/HD * CC, /*tile_d1=*/3,
                    /*tensor_d0=*/CC * CC, /*tensor_d1=*/3,
                    /*stride0=*/(unsigned long long)CC * CC);
        tdm_wait();
#else
        for (int s = 0; s < 3; ++s)
            for (int idx = lane; idx < HD * CC; idx += 32)
                ldsw[s * HD * CC + idx] = wqkv[(size_t)h * HD * CC + (size_t)s * CC * CC + idx];
#endif
    }
    __syncthreads();

    const int aOff = hseg * 16;             // A-tile K sub-offset (contiguous)
    const int bOff = hseg * 16;             // B-tile K sub-offset (contiguous)

    v8f aq0[2] = {}, aq1[2] = {}, ak0[2] = {}, ak1[2] = {}, av0[2] = {}, av1[2] = {};

#pragma unroll
    for (int kb = 0; kb < CC; kb += 32) {
        v16bf a[2];
#pragma unroll
        for (int u = 0; u < 2; ++u) {
            const __bf16* arow = xp + (size_t)(m0 + u * 16 + col) * CC;
            v8bf lo = *(const v8bf*)(arow + kb + aOff);
            v8bf hi = *(const v8bf*)(arow + kb + aOff + 8);
#pragma unroll
            for (int i = 0; i < 8; ++i) { a[u][i] = lo[i]; a[u][i + 8] = hi[i]; }
        }
        const __bf16* wl = ldsw + col * CC + kb + bOff;
        // --- q segment ---
        {
            v16bf b0 = *(const v16bf*)(wl);
            v16bf b1 = *(const v16bf*)(wl + 16 * CC);
#pragma unroll
            for (int u = 0; u < 2; ++u) {
                aq0[u] = __builtin_amdgcn_wmma_f32_16x16x32_bf16(false, a[u], false, b0, (short)0, aq0[u], false, false);
                aq1[u] = __builtin_amdgcn_wmma_f32_16x16x32_bf16(false, a[u], false, b1, (short)0, aq1[u], false, false);
            }
        }
        // --- k segment ---
        {
            v16bf b0 = *(const v16bf*)(wl + HD * CC);
            v16bf b1 = *(const v16bf*)(wl + HD * CC + 16 * CC);
#pragma unroll
            for (int u = 0; u < 2; ++u) {
                ak0[u] = __builtin_amdgcn_wmma_f32_16x16x32_bf16(false, a[u], false, b0, (short)0, ak0[u], false, false);
                ak1[u] = __builtin_amdgcn_wmma_f32_16x16x32_bf16(false, a[u], false, b1, (short)0, ak1[u], false, false);
            }
        }
        // --- v segment ---
        {
            v16bf b0 = *(const v16bf*)(wl + 2 * HD * CC);
            v16bf b1 = *(const v16bf*)(wl + 2 * HD * CC + 16 * CC);
#pragma unroll
            for (int u = 0; u < 2; ++u) {
                av0[u] = __builtin_amdgcn_wmma_f32_16x16x32_bf16(false, a[u], false, b0, (short)0, av0[u], false, false);
                av1[u] = __builtin_amdgcn_wmma_f32_16x16x32_bf16(false, a[u], false, b1, (short)0, av1[u], false, false);
            }
        }
    }

    const int nq = h * HD, nk = CC + h * HD, nv = 2 * CC + h * HD;
    const float bq0s = qkv_b[nq + col], bq1s = qkv_b[nq + 16 + col];
    const float bk0s = qkv_b[nk + col], bk1s = qkv_b[nk + 16 + col];
    const float bv0s = qkv_b[nv + col], bv1s = qkv_b[nv + 16 + col];
    const float scale = 0.17677669529663687f;  // 32^-0.5

#pragma unroll
    for (int u = 0; u < 2; ++u) {
        const int mb = m0 + u * 16;
#pragma unroll
        for (int r = 0; r < 8; ++r) {
            aq0[u][r] += bq0s; aq1[u][r] += bq1s;
            ak0[u][r] += bk0s; ak1[u][r] += bk1s;
            const size_t m = (size_t)(mb + r + 8 * hseg);
            v_ws[m * CC + h * HD + col]      = (__bf16)(av0[u][r] + bv0s);
            v_ws[m * CC + h * HD + 16 + col] = (__bf16)(av1[u][r] + bv1s);
        }
        // dot = scale * sum_d q*k; reduce across the 16 lanes of each half
#pragma unroll
        for (int r = 0; r < 8; ++r) {
            float p = aq0[u][r] * ak0[u][r] + aq1[u][r] * ak1[u][r];
            p += __shfl_xor(p, 1, 32);
            p += __shfl_xor(p, 2, 32);
            p += __shfl_xor(p, 4, 32);
            p += __shfl_xor(p, 8, 32);
            if (col == 0) {
                const int m  = mb + r + 8 * hseg;
                const int b  = m / PIX;
                const int pp = m - b * PIX;
                dot_ws[((size_t)(b * NHD + h)) * PIX + pp] = p * scale;
            }
        }
    }
}

// ---------------------------------------------------------------------------
// 4) Neighborhood softmax + weighted v sum. Wave = (pixel, head); lane = dim.
// ---------------------------------------------------------------------------
__global__ void natten_kernel(const float* __restrict__ dot_ws,
                              const float* __restrict__ rpb,
                              const __bf16* __restrict__ v_ws,
                              __bf16* __restrict__ hidden) {
    const int lane = threadIdx.x & 31;
    const int wave = (blockIdx.x * blockDim.x + threadIdx.x) >> 5;
    const int x = wave % WID;
    int t = wave / WID;
    const int y = t % HHT;  t /= HHT;
    const int h = t % NHD;
    const int b = t / NHD;

    const float* dp = dot_ws + ((size_t)(b * NHD + h)) * PIX;
    const float* rp = rpb + h * KS * KS;

    float lg[KS * KS];
    float mx = -3.4e38f;
#pragma unroll
    for (int i = 0; i < KS; ++i) {
        int yy = y + i; if (yy >= HHT) yy -= HHT;
#pragma unroll
        for (int j = 0; j < KS; ++j) {
            int xx = x + j; if (xx >= WID) xx -= WID;
            float l = dp[yy * WID + xx] + rp[i * KS + j];
            lg[i * KS + j] = l;
            mx = fmaxf(mx, l);
        }
    }
    float s = 0.f;
#pragma unroll
    for (int q = 0; q < KS * KS; ++q) { lg[q] = __expf(lg[q] - mx); s += lg[q]; }
    const float inv = 1.0f / s;

    float acc = 0.f;
#pragma unroll
    for (int i = 0; i < KS; ++i) {
        int yy = y + i; if (yy >= HHT) yy -= HHT;
#pragma unroll
        for (int j = 0; j < KS; ++j) {
            int xx = x + j; if (xx >= WID) xx -= WID;
            const size_t nm = (size_t)b * PIX + yy * WID + xx;
            acc += lg[i * KS + j] * inv * (float)v_ws[nm * CC + h * HD + lane];
        }
    }
    hidden[((size_t)b * PIX + y * WID + x) * CC + h * HD + lane] = (__bf16)acc;
}

// ---------------------------------------------------------------------------
// 5) proj GEMM + bias, scatter-store to (B,C,H,W).
//    Block = 32 M-rows x full N; proj weights (32KB) staged via TDM;
//    wave = 2 M-subtiles x 1 N-tile.
// ---------------------------------------------------------------------------
__global__ void __launch_bounds__(256, 1)
proj_kernel(const __bf16* __restrict__ hidden,
            const __bf16* __restrict__ wproj,
            const float* __restrict__ proj_b,
            float* __restrict__ out) {
    __shared__ __align__(64) __bf16 ldsw[CC * CC];

    const int lane = threadIdx.x & 31;
    const int nt   = threadIdx.x >> 5;      // 8 waves = 8 N-tiles
    const int m0   = blockIdx.x * 32;
    const int n0   = nt * 16;
    const int col  = lane & 15;
    const int hseg = lane >> 4;
    const int aOff = hseg * 16;
    const int bOff = hseg * 16;

    if (nt == 0) {
#ifdef HAVE_TDM
        tdm_load_2d((unsigned)(uintptr_t)ldsw, wproj,
                    /*tile_d0=*/CC * CC, /*tile_d1=*/1,
                    /*tensor_d0=*/CC * CC, /*tensor_d1=*/1,
                    /*stride0=*/(unsigned long long)CC * CC);
        tdm_wait();
#else
        for (int idx = lane; idx < CC * CC; idx += 32) ldsw[idx] = wproj[idx];
#endif
    }
    __syncthreads();

    v8f acc[2] = {};
#pragma unroll
    for (int kb = 0; kb < CC; kb += 32) {
        v16bf a[2];
#pragma unroll
        for (int u = 0; u < 2; ++u) {
            const __bf16* arow = hidden + (size_t)(m0 + u * 16 + col) * CC;
            v8bf lo = *(const v8bf*)(arow + kb + aOff);
            v8bf hi = *(const v8bf*)(arow + kb + aOff + 8);
#pragma unroll
            for (int i = 0; i < 8; ++i) { a[u][i] = lo[i]; a[u][i + 8] = hi[i]; }
        }
        v16bf bm = *(const v16bf*)(ldsw + (n0 + col) * CC + kb + bOff);
#pragma unroll
        for (int u = 0; u < 2; ++u)
            acc[u] = __builtin_amdgcn_wmma_f32_16x16x32_bf16(false, a[u], false, bm, (short)0, acc[u], false, false);
    }

    const float bias = proj_b[n0 + col];
#pragma unroll
    for (int u = 0; u < 2; ++u) {
#pragma unroll
        for (int r = 0; r < 8; ++r) {
            const int m  = m0 + u * 16 + r + 8 * hseg;
            const int b  = m / PIX;
            const int pp = m - b * PIX;
            out[((size_t)(b * CC + n0 + col)) * PIX + pp] = acc[u][r] + bias;
        }
    }
}

// ---------------------------------------------------------------------------
extern "C" void kernel_launch(void* const* d_in, const int* in_sizes, int n_in,
                              void* d_out, int out_size, void* d_ws, size_t ws_size,
                              hipStream_t stream) {
    const float* x      = (const float*)d_in[0];
    const float* qkv_w  = (const float*)d_in[1];
    const float* qkv_b  = (const float*)d_in[2];
    const float* rpb    = (const float*)d_in[3];
    const float* proj_w = (const float*)d_in[4];
    const float* proj_b = (const float*)d_in[5];
    float* out = (float*)d_out;

    char* ws = (char*)d_ws;
    size_t off = 0;
    auto carve = [&](size_t bytes) -> char* {
        char* p = ws + off;
        off += (bytes + 255) & ~(size_t)255;
        return p;
    };
    __bf16* xp     = (__bf16*)carve((size_t)MROWS * CC * 2);
    __bf16* wqkv   = (__bf16*)carve((size_t)3 * CC * CC * 2);
    __bf16* wproj  = (__bf16*)carve((size_t)CC * CC * 2);
    __bf16* v_ws   = (__bf16*)carve((size_t)MROWS * CC * 2);
    float*  dot_ws = (float*) carve((size_t)BB * NHD * PIX * 4);
    __bf16* hidden = (__bf16*)carve((size_t)MROWS * CC * 2);
    (void)ws_size; (void)in_sizes; (void)n_in; (void)out_size;

    transpose_x_kernel<<<BB * (CC / 32) * (PIX / 32), 256, 0, stream>>>(x, xp);
    convert_w_kernel<<<(3 * CC * CC + CC * CC) / 256, 256, 0, stream>>>(qkv_w, proj_w, wqkv, wproj);
    // 3136 (32-row M tiles) / 8 waves * 4 heads
    qkv_dot_v_kernel<<<(MROWS / 32 / 8) * NHD, 256, 0, stream>>>(xp, wqkv, qkv_b, v_ws, dot_ws);
    natten_kernel<<<(size_t)BB * PIX * NHD / 8, 256, 0, stream>>>(dot_ws, rpb, v_ws, hidden);
    proj_kernel<<<MROWS / 32, 256, 0, stream>>>(hidden, wproj, proj_b, out);
}
